// VectorQuantizer_41532333753121
// MI455X (gfx1250) — compile-verified, matrix operand order checked by reading, not run on hardware
//
#include <hip/hip_runtime.h>
#include <math.h>

// Problem constants
#define DIM 256
#define LSEQ 2048
#define NB 32
#define NROWS 65536          // NB*LSEQ
#define KCODES 1024
#define NTILES 64            // KCODES/16
#define QELEMS 16777216      // NB*DIM*LSEQ

// Workspace layout (dword offsets)
#define WSB_OFF 0            // 64*8*32*8 = 131072 dwords: f16 B fragments
#define ENORM_OFF 131072     // 1024 floats
#define IDX_OFF 132096       // 65536 ints
#define CNT_OFF 197632       // 1024 uints
#define PART_OFF 198656      // 65536 floats (per-block loss partials)

typedef _Float16 v16h __attribute__((ext_vector_type(16)));
typedef float v8f __attribute__((ext_vector_type(8)));
typedef int v4i __attribute__((ext_vector_type(4)));

union Frag { v16h h; unsigned u[8]; uint4 q[2]; };

// 16-bit A/B 16x32 fragment K index for (lane, vgpr) per CDNA5 ISA 7.12.2
__device__ __forceinline__ int kb(int lane, int v) {
  return 2 * v + ((v & 4) ? 8 : 0) + ((lane & 16) ? 8 : 0);
}

__device__ __forceinline__ unsigned pkh2(float a, float b) {
  union { _Float16 h[2]; unsigned u; } r;
  r.h[0] = (_Float16)a; r.h[1] = (_Float16)b;
  return r.u;
}

#if defined(__gfx1250__) && __has_builtin(__builtin_amdgcn_global_load_async_to_lds_b128)
#define USE_ASYNC 1
#else
#define USE_ASYNC 0
#endif

__device__ __forceinline__ void stage8k(const unsigned* __restrict__ src,
                                        unsigned* dst, int tid) {
  // copy 8192 bytes with 256 threads, 32B per thread
#if USE_ASYNC
  char* s = (char*)src + tid * 32;  // builtin params are non-const generic int4*
  char* d = (char*)dst + tid * 32;
  __builtin_amdgcn_global_load_async_to_lds_b128((v4i*)s, (v4i*)d, 0, 0);
  __builtin_amdgcn_global_load_async_to_lds_b128((v4i*)(s + 16), (v4i*)(d + 16), 0, 0);
#else
  const uint4* s4 = (const uint4*)src;
  uint4* d4 = (uint4*)dst;
  d4[2 * tid] = s4[2 * tid];
  d4[2 * tid + 1] = s4[2 * tid + 1];
#endif
}

__device__ __forceinline__ void wait_async0() {
#if USE_ASYNC
#if __has_builtin(__builtin_amdgcn_s_wait_asynccnt)
  __builtin_amdgcn_s_wait_asynccnt(0);
#else
  asm volatile("s_wait_asynccnt 0" ::: "memory");
#endif
#endif
}

// ---------------- Kernel: zero the code-usage histogram ----------------
__global__ void zero_counts(unsigned* cnt) {
  int i = blockIdx.x * 256 + threadIdx.x;
  if (i < KCODES) cnt[i] = 0;
}

// ---------------- Kernel: ||e_k||^2 ----------------
__global__ __launch_bounds__(256) void enorm_kernel(const float* __restrict__ emb,
                                                    float* __restrict__ enorm) {
  __shared__ float red[256];
  int k = blockIdx.x, d = threadIdx.x;
  float v = emb[k * DIM + d];
  red[d] = v * v;
  __syncthreads();
  for (int s = 128; s > 0; s >>= 1) {
    if (d < s) red[d] += red[d + s];
    __syncthreads();
  }
  if (d == 0) enorm[k] = red[0];
}

// ---------------- Kernel: embedding -> f16 B fragments ----------------
// Layout: wsB[((tile*8 + dt)*32 + lane)*8 + v]; B is 32x16 (K x N), N = lane&15
__global__ __launch_bounds__(256) void prep_b(const float* __restrict__ emb,
                                              unsigned* __restrict__ wsB) {
  int tile = blockIdx.x;
  int tid = threadIdx.x, dt = tid >> 5, lane = tid & 31;
  int code = tile * 16 + (lane & 15);
  const float* er = emb + (size_t)code * DIM;
#pragma unroll
  for (int v = 0; v < 8; ++v) {
    int d = dt * 32 + kb(lane, v);
    wsB[(((size_t)tile * 8 + dt) * 32 + lane) * 8 + v] = pkh2(er[d], er[d + 1]);
  }
}

// ---------------- Kernel: WMMA distance + argmin ----------------
// 512 blocks x 256 threads (8 waves); wave w owns rows n0..n0+15
__global__ __launch_bounds__(256) void vq_main(const float* __restrict__ x,
                                               const unsigned* __restrict__ wsB,
                                               const float* __restrict__ enorm,
                                               int* __restrict__ indices) {
  __shared__ unsigned ldsB[2][2048];  // two 8KB B-tile buffers
  __shared__ float ldsE[KCODES];
  int tid = threadIdx.x, wave = tid >> 5, lane = tid & 31, m = lane & 15;

  for (int i = tid; i < KCODES; i += 256) ldsE[i] = enorm[i];

  // Load this wave's A fragments (16 rows x 256 dims, f32 -> f16)
  int n0 = blockIdx.x * 128 + wave * 16;
  int bb = n0 >> 11, l0 = n0 & 2047;
  const float* xb = x + (size_t)bb * DIM * LSEQ + l0 + m;  // x[b][d][l0+m] at xb[d*2048]
  Frag a[8];
#pragma unroll
  for (int dt = 0; dt < 8; ++dt)
#pragma unroll
    for (int v = 0; v < 8; ++v) {
      int d = dt * 32 + kb(lane, v);
      a[dt].u[v] = pkh2(xb[(size_t)d * LSEQ], xb[(size_t)(d + 1) * LSEQ]);
    }

  float best_d[8];
  int best_i[8];
#pragma unroll
  for (int j = 0; j < 8; ++j) { best_d[j] = __builtin_inff(); best_i[j] = 0; }

  stage8k(wsB, ldsB[0], tid);
  wait_async0();
  __syncthreads();

  for (int ct = 0; ct < NTILES; ++ct) {
    int cur = ct & 1;
    if (ct + 1 < NTILES) stage8k(wsB + (size_t)(ct + 1) * 2048, ldsB[1 - cur], tid);

    v8f acc = {};
#pragma unroll
    for (int dt = 0; dt < 8; ++dt) {
      Frag bf;
      const uint4* p = (const uint4*)(&ldsB[cur][((dt * 32 + lane) << 3)]);
      bf.q[0] = p[0];
      bf.q[1] = p[1];
      acc = __builtin_amdgcn_wmma_f32_16x16x32_f16(false, a[dt].h, false, bf.h,
                                                   (short)0, acc, false, false);
    }
    // distance = ||e||^2 - 2*score  (||x||^2 constant per row, dropped)
    float en = ldsE[ct * 16 + m];
#pragma unroll
    for (int j = 0; j < 8; ++j) {
      float dist = en - 2.0f * acc[j];
      if (dist < best_d[j]) { best_d[j] = dist; best_i[j] = ct * 16 + m; }
    }
    if (ct + 1 < NTILES) { wait_async0(); __syncthreads(); }
  }

  // Cross-lane argmin over N (16 lanes per half); C/D layout: M = j + 8*(lane>=16)
#pragma unroll
  for (int j = 0; j < 8; ++j) {
    float d = best_d[j];
    int ii = best_i[j];
    for (int off = 1; off < 16; off <<= 1) {
      float od = __shfl_xor(d, off, 32);
      int oi = __shfl_xor(ii, off, 32);
      if (od < d || (od == d && oi < ii)) { d = od; ii = oi; }
    }
    if (m == 0) indices[n0 + j + ((lane & 16) ? 8 : 0)] = ii;
  }
}

// ---------------- Kernel: gather quantized (b,c,l) + MSE partials ----------------
__global__ __launch_bounds__(256) void quant_loss(const float* __restrict__ x,
                                                  const float* __restrict__ emb,
                                                  const int* __restrict__ indices,
                                                  float* __restrict__ out_q,
                                                  float* __restrict__ part) {
  __shared__ float red[256];
  size_t g = (size_t)blockIdx.x * 256 + threadIdx.x;  // flat (b, d, l)
  int l = (int)(g & 2047);
  int d = (int)((g >> 11) & 255);
  int bb = (int)(g >> 19);
  int n = bb * LSEQ + l;
  int idx = indices[n];
  float q = emb[(size_t)idx * DIM + d];
  out_q[g] = q;
  float diff = q - x[g];
  red[threadIdx.x] = diff * diff;
  __syncthreads();
  for (int s = 128; s > 0; s >>= 1) {
    if (threadIdx.x < s) red[threadIdx.x] += red[threadIdx.x + s];
    __syncthreads();
  }
  if (threadIdx.x == 0) part[blockIdx.x] = red[0];
}

// ---------------- Kernel: code-usage histogram ----------------
__global__ void count_kernel(const int* __restrict__ indices, unsigned* __restrict__ cnt) {
  int i = blockIdx.x * 256 + threadIdx.x;
  atomicAdd(&cnt[indices[i]], 1u);
}

// ---------------- Kernel: one-hot encodings (N, K) ----------------
__global__ __launch_bounds__(256) void enc_kernel(const int* __restrict__ indices,
                                                  float* __restrict__ enc) {
  size_t base = ((size_t)blockIdx.x * 256 + threadIdx.x) * 4;
  int n = (int)(base >> 10);
  int k0 = (int)(base & 1023);
  int id = indices[n];
  float2* p = (float2*)(enc + base);  // enc is 8B-aligned (even float offset in d_out)
  p[0] = make_float2(k0 == id ? 1.f : 0.f, k0 + 1 == id ? 1.f : 0.f);
  p[1] = make_float2(k0 + 2 == id ? 1.f : 0.f, k0 + 3 == id ? 1.f : 0.f);
}

// ---------------- Kernel: loss + perplexity ----------------
__global__ __launch_bounds__(256) void finalize_kernel(const float* __restrict__ part,
                                                       const unsigned* __restrict__ cnt,
                                                       float* __restrict__ out_loss,
                                                       float* __restrict__ out_perp) {
  __shared__ float red[256];
  int t = threadIdx.x;
  float s = 0.f;
  for (int i = t; i < NROWS; i += 256) s += part[i];
  red[t] = s;
  __syncthreads();
  for (int k = 128; k > 0; k >>= 1) {
    if (t < k) red[t] += red[t + k];
    __syncthreads();
  }
  if (t == 0) *out_loss = 1.25f * red[0] / (float)QELEMS;  // q_loss + 0.25*e_loss
  __syncthreads();
  float e = 0.f;
  for (int k = t; k < KCODES; k += 256) {
    float avg = (float)cnt[k] * (1.0f / (float)NROWS);
    e += avg * logf(avg + 1e-10f);
  }
  red[t] = e;
  __syncthreads();
  for (int k = 128; k > 0; k >>= 1) {
    if (t < k) red[t] += red[t + k];
    __syncthreads();
  }
  if (t == 0) *out_perp = expf(-red[0]);
}

extern "C" void kernel_launch(void* const* d_in, const int* in_sizes, int n_in,
                              void* d_out, int out_size, void* d_ws, size_t ws_size,
                              hipStream_t stream) {
  const float* x = (const float*)d_in[0];    // (32, 256, 2048)
  const float* emb = (const float*)d_in[1];  // (1024, 256)
  float* out = (float*)d_out;
  unsigned* ws = (unsigned*)d_ws;  // needs ~1.01 MB

  unsigned* wsB = ws + WSB_OFF;
  float* enorm = (float*)(ws + ENORM_OFF);
  int* idx = (int*)(ws + IDX_OFF);
  unsigned* cnt = ws + CNT_OFF;
  float* part = (float*)(ws + PART_OFF);

  float* out_loss = out;                     // [0]
  float* out_q = out + 1;                    // [1 .. 1+16777216)
  float* out_perp = out + 1 + QELEMS;        // [16777217]
  float* out_enc = out + 2 + QELEMS;         // [16777218 ..)

  zero_counts<<<4, 256, 0, stream>>>(cnt);
  enorm_kernel<<<KCODES, 256, 0, stream>>>(emb, enorm);
  prep_b<<<NTILES, 256, 0, stream>>>(emb, wsB);
  vq_main<<<NROWS / 128, 256, 0, stream>>>(x, wsB, enorm, idx);
  quant_loss<<<QELEMS / 256, 256, 0, stream>>>(x, emb, idx, out_q, part);
  count_kernel<<<NROWS / 256, 256, 0, stream>>>(idx, cnt);
  enc_kernel<<<(NROWS * (KCODES / 4)) / 256, 256, 0, stream>>>(idx, out_enc);
  finalize_kernel<<<1, 256, 0, stream>>>(part, cnt, out_loss, out_perp);
}
